// GeneralRNN_12472585027810
// MI455X (gfx1250) — matrix operand order repaired and checked by reading, not compiled, not run on hardware
//
#include <hip/hip_runtime.h>

// ---------------------------------------------------------------------------
// Persistent bf16-WMMA RNN for MI455X (gfx1250, wave32).
//   B=256, S=1024, D_IN=64, H=256, W_H=512, W_Y=256, D_OUT=64
// 8 workgroups x 256 threads (8 waves). Each workgroup owns 32 batch rows
// (2 WMMA M-tiles) and runs the full 1024-step recurrence with the hidden
// state resident in LDS. Weights are pre-swizzled to bf16 WMMA B-fragment
// layout in d_ws and streamed from L2 each step.
//
// Round-1 fix: cap kt-loop unrolling at 2. Full unroll made the compiler keep
// ~12 B-fragments live (>256 VGPRs -> s_set_vgpr_msb churn + scratch spills
// in the inner loop). Unroll-2 keeps ~160 VGPRs live: no spills, no MSB ops,
// and still software-pipelines next-iteration global loads under the WMMAs.
// ---------------------------------------------------------------------------

typedef __attribute__((ext_vector_type(16))) __bf16 v16bf;
typedef __attribute__((ext_vector_type(8)))  __bf16 v8bf;
typedef __attribute__((ext_vector_type(8)))  float  v8f;
typedef __attribute__((ext_vector_type(4)))  float  v4f;

#define B_SZ  256
#define S_SZ  1024
#define H_SZ  256
#define DIN   64
#define WH    512
#define WY    256
#define DOUT  64

__device__ __forceinline__ v8f wmma_bf16(v16bf a, v16bf b, v8f c) {
  // D = A(16x32 bf16) * B(32x16 bf16) + C(16x16 f32)
  return __builtin_amdgcn_wmma_f32_16x16x32_bf16(false, a, false, b,
                                                 (short)0, c, false, false);
}

__device__ __forceinline__ float silu_f(float v) {
  return v / (1.0f + __expf(-v));
}

// ---------------------------------------------------------------------------
// Pre-swizzle: fp32 row-major W[K,N] -> bf16 WMMA B-fragments.
// Fragment (ntile,ktile) is 512 contiguous bf16 (1KB): index
//   ((ntile*KT + ktile)*32 + lane)*16 + elem
// with  n = ntile*16 + (lane&15),  k = ktile*32 + 8*(lane>>4) + (elem&7) + 16*(elem>>3).
// ---------------------------------------------------------------------------
__global__ void swizzle_w(const float* __restrict__ src,
                          unsigned short* __restrict__ dst_, int K, int N) {
  __bf16* dst = (__bf16*)dst_;
  const int KT = K >> 5;
  const int total = K * N;
  for (int idx = blockIdx.x * blockDim.x + threadIdx.x; idx < total;
       idx += gridDim.x * blockDim.x) {
    const int frag   = idx >> 9;
    const int within = idx & 511;
    const int lane   = within >> 4;
    const int elem   = within & 15;
    const int ktile  = frag % KT;
    const int ntile  = frag / KT;
    const int k = ktile * 32 + 8 * (lane >> 4) + (elem & 7) + 16 * (elem >> 3);
    const int n = ntile * 16 + (lane & 15);
    dst[idx] = (__bf16)src[k * N + n];
  }
}

// ---------------------------------------------------------------------------
// Persistent RNN kernel.
// LDS (dynamic, 53248 B):
//   sHX : [2 mt][10 kt][512] bf16  A-frags of hx  (kt 0..7 = h, kt 8..9 = x)
//   sA1 : [2 mt][16 kt][512] bf16  A-frags of silu(a1)        (aliases sA2)
//   sA2 : [2 mt][ 8 kt][512] bf16  A-frags of silu(a2)
// ---------------------------------------------------------------------------
__global__ __launch_bounds__(256) void rnn_persistent(
    const float* __restrict__ x,
    const unsigned short* __restrict__ w1h_, const unsigned short* __restrict__ w2h_,
    const unsigned short* __restrict__ w1y_, const unsigned short* __restrict__ w2y_,
    const float* __restrict__ b1h, const float* __restrict__ b2h,
    const float* __restrict__ b1y, const float* __restrict__ b2y,
    float* __restrict__ outs, float* __restrict__ hid) {
  const __bf16* w1h = (const __bf16*)w1h_;
  const __bf16* w2h = (const __bf16*)w2h_;
  const __bf16* w1y = (const __bf16*)w1y_;
  const __bf16* w2y = (const __bf16*)w2y_;

  extern __shared__ __bf16 smem[];
  __bf16* sHX = smem;           // 10240 elems
  __bf16* sA1 = smem + 10240;   // 16384 elems
  __bf16* sA2 = sA1;            // first 8192 elems, disjoint lifetime

  const int tid  = threadIdx.x;
  const int wave = tid >> 5;
  const int lane = tid & 31;
  const int b0   = blockIdx.x * 32;   // this WG's first batch row
  const int nl   = lane & 15;

  // Biases stay in registers for the whole sequence.
  float rb1h[4], rb2h[2], rb1y[2];
#pragma unroll
  for (int i = 0; i < 4; ++i) rb1h[i] = b1h[(wave * 4 + i) * 16 + nl];
#pragma unroll
  for (int i = 0; i < 2; ++i) rb2h[i] = b2h[(wave * 2 + i) * 16 + nl];
#pragma unroll
  for (int i = 0; i < 2; ++i) rb1y[i] = b1y[(wave * 2 + i) * 16 + nl];
  const float rb2y = (wave < 4) ? b2y[wave * 16 + nl] : 0.0f;

  // D-layout -> A-layout scatter constants (per lane).
  // D elem r in this lane is matrix element (m = r + 8*(lane>>4), n = lane&15);
  // as next-GEMM A element (m, k=ntile*16+n) it lives at
  //   lane' = m + 16*((n>>3)&1),  elem' = (n&7) | ((ntile&1)<<3),  kt' = ntile>>1.
  const int mhi       = lane >> 4;
  const int lane2base = 8 * mhi + 16 * ((lane >> 3) & 1);
  const int e2lo      = lane & 7;

  // x_t staging decomposition: 2048 bf16 elems / 256 threads = 8 each,
  // arranged so each thread reads 8 consecutive floats of x (2x v4f).
  const int xs_mt   = tid >> 7;
  const int xs_ktl  = (tid >> 6) & 1;
  const int xs_lane = (tid >> 1) & 31;
  const int xs_e0   = (tid & 1) * 8;
  const int xs_k0   = xs_ktl * 32 + 8 * (xs_lane >> 4) + 16 * (tid & 1);
  const int xs_b    = b0 + xs_mt * 16 + (xs_lane & 15);
  const float* xs_src = x + (size_t)xs_b * S_SZ * DIN + xs_k0;
  __bf16* xs_dst = &sHX[((xs_mt * 10 + 8 + xs_ktl) * 32 + xs_lane) * 16 + xs_e0];

  // h0 = 0: clear hidden-state fragments and hiddens[0].
  {
    unsigned int* z = (unsigned int*)sHX;
    for (int i = tid; i < 5120; i += 256) z[i] = 0u;
    for (int i = tid; i < 32 * H_SZ; i += 256) hid[(size_t)b0 * H_SZ + i] = 0.0f;
  }
  __syncthreads();

  for (int t = 0; t < S_SZ; ++t) {
    // ---- stage x_t into bf16 A-fragment layout (sHX kt 8..9) ----
    {
      const float* sp = xs_src + (size_t)t * DIN;
      v4f f0 = *(const v4f*)sp;
      v4f f1 = *(const v4f*)(sp + 4);
      v8bf pk;
#pragma unroll
      for (int i = 0; i < 4; ++i) { pk[i] = (__bf16)f0[i]; pk[4 + i] = (__bf16)f1[i]; }
      *(v8bf*)xs_dst = pk;
    }
    __syncthreads();  // x staged; h_new from previous step already in sHX

    // ---- GEMM1: [32,320] x [320,512], wave owns 4 N-tiles, 2 M-tiles ----
    v8f c1[2][4] = {};
#pragma unroll 2
    for (int kt = 0; kt < 10; ++kt) {
      v16bf a0 = *(const v16bf*)&sHX[((0 * 10 + kt) * 32 + lane) * 16];
      v16bf a1 = *(const v16bf*)&sHX[((1 * 10 + kt) * 32 + lane) * 16];
#pragma unroll
      for (int ntl = 0; ntl < 4; ++ntl) {
        const int nt = wave * 4 + ntl;
        v16bf bb = *(const v16bf*)&w1h[(nt * 10 + kt) * 512 + lane * 16];
        c1[0][ntl] = wmma_bf16(a0, bb, c1[0][ntl]);
        c1[1][ntl] = wmma_bf16(a1, bb, c1[1][ntl]);
      }
    }
    // epilogue: +b1h, SiLU, bf16 -> sA1 A-frags
#pragma unroll
    for (int mt = 0; mt < 2; ++mt) {
#pragma unroll
      for (int ntl = 0; ntl < 4; ++ntl) {
        const int nt  = wave * 4 + ntl;
        const int kt2 = nt >> 1;
        __bf16* dst = &sA1[((mt * 16 + kt2) * 32 + lane2base) * 16 +
                           (e2lo | ((nt & 1) << 3))];
        v8f c = c1[mt][ntl];
#pragma unroll
        for (int r = 0; r < 8; ++r) dst[r * 16] = (__bf16)silu_f(c[r] + rb1h[ntl]);
      }
    }
    __syncthreads();

    // ---- GEMM2: [32,512] x [512,256] -> h_new, wave owns 2 N-tiles ----
    v8f c2[2][2] = {};
#pragma unroll 2
    for (int kt = 0; kt < 16; ++kt) {
      v16bf a0 = *(const v16bf*)&sA1[((0 * 16 + kt) * 32 + lane) * 16];
      v16bf a1 = *(const v16bf*)&sA1[((1 * 16 + kt) * 32 + lane) * 16];
#pragma unroll
      for (int ntl = 0; ntl < 2; ++ntl) {
        const int nt = wave * 2 + ntl;
        v16bf bb = *(const v16bf*)&w2h[(nt * 16 + kt) * 512 + lane * 16];
        c2[0][ntl] = wmma_bf16(a0, bb, c2[0][ntl]);
        c2[1][ntl] = wmma_bf16(a1, bb, c2[1][ntl]);
      }
    }
    // epilogue: +b2h; fp32 -> hiddens[t+1]; bf16 -> sHX (state for GEMM3/next step)
#pragma unroll
    for (int mt = 0; mt < 2; ++mt) {
#pragma unroll
      for (int ntl = 0; ntl < 2; ++ntl) {
        const int nt  = wave * 2 + ntl;
        const int kt2 = nt >> 1;
        __bf16* dst = &sHX[((mt * 10 + kt2) * 32 + lane2base) * 16 +
                           (e2lo | ((nt & 1) << 3))];
        float* gdst = &hid[(size_t)(t + 1) * B_SZ * H_SZ +
                           (size_t)(b0 + mt * 16) * H_SZ + nt * 16 + nl];
        v8f c = c2[mt][ntl];
#pragma unroll
        for (int r = 0; r < 8; ++r) {
          const float h = c[r] + rb2h[ntl];
          gdst[(size_t)(r + 8 * mhi) * H_SZ] = h;
          dst[r * 16] = (__bf16)h;
        }
      }
    }
    __syncthreads();

    // ---- GEMM3: [32,256] x [256,256], wave owns 2 N-tiles ----
    v8f c3[2][2] = {};
#pragma unroll 2
    for (int kt = 0; kt < 8; ++kt) {
      v16bf a0 = *(const v16bf*)&sHX[((0 * 10 + kt) * 32 + lane) * 16];
      v16bf a1 = *(const v16bf*)&sHX[((1 * 10 + kt) * 32 + lane) * 16];
#pragma unroll
      for (int ntl = 0; ntl < 2; ++ntl) {
        const int nt = wave * 2 + ntl;
        v16bf bb = *(const v16bf*)&w1y[(nt * 8 + kt) * 512 + lane * 16];
        c3[0][ntl] = wmma_bf16(a0, bb, c3[0][ntl]);
        c3[1][ntl] = wmma_bf16(a1, bb, c3[1][ntl]);
      }
    }
    // epilogue: +b1y, SiLU, bf16 -> sA2 A-frags
#pragma unroll
    for (int mt = 0; mt < 2; ++mt) {
#pragma unroll
      for (int ntl = 0; ntl < 2; ++ntl) {
        const int nt  = wave * 2 + ntl;
        const int kt2 = nt >> 1;
        __bf16* dst = &sA2[((mt * 8 + kt2) * 32 + lane2base) * 16 +
                           (e2lo | ((nt & 1) << 3))];
        v8f c = c3[mt][ntl];
#pragma unroll
        for (int r = 0; r < 8; ++r) dst[r * 16] = (__bf16)silu_f(c[r] + rb1y[ntl]);
      }
    }
    __syncthreads();

    // ---- GEMM4: [32,256] x [256,64] -> y, waves 0..3 take 1 N-tile each ----
    if (wave < 4) {  // wave-uniform: EXEC stays all-ones for WMMA
      v8f c4[2] = {};
#pragma unroll 2
      for (int kt = 0; kt < 8; ++kt) {
        v16bf a0 = *(const v16bf*)&sA2[((0 * 8 + kt) * 32 + lane) * 16];
        v16bf a1 = *(const v16bf*)&sA2[((1 * 8 + kt) * 32 + lane) * 16];
        v16bf bb = *(const v16bf*)&w2y[(wave * 8 + kt) * 512 + lane * 16];
        c4[0] = wmma_bf16(a0, bb, c4[0]);
        c4[1] = wmma_bf16(a1, bb, c4[1]);
      }
#pragma unroll
      for (int mt = 0; mt < 2; ++mt) {
        float* ydst = &outs[((size_t)(b0 + mt * 16) * S_SZ + t) * DOUT +
                            wave * 16 + nl];
        v8f c = c4[mt];
#pragma unroll
        for (int r = 0; r < 8; ++r)
          ydst[(size_t)(r + 8 * mhi) * S_SZ * DOUT] = c[r] + rb2y;
      }
    }
    // No trailing barrier needed: next iteration's x staging touches only
    // sHX kt 8..9 (consumed before this step's B2), and the post-x barrier
    // orders everything else.
  }
}

// ---------------------------------------------------------------------------
// Host launcher.
// d_in: x, W1h, b1h, W2h, b2h, W1y, b1y, W2y, b2y (all fp32)
// d_out: outs [256,1024,64] fp32 then hiddens [1025,256,256] fp32
// d_ws : bf16 swizzled weights (753,664 bytes)
// ---------------------------------------------------------------------------
extern "C" void kernel_launch(void* const* d_in, const int* in_sizes, int n_in,
                              void* d_out, int out_size, void* d_ws, size_t ws_size,
                              hipStream_t stream) {
  (void)in_sizes; (void)n_in; (void)out_size; (void)ws_size;

  const float* x   = (const float*)d_in[0];
  const float* W1h = (const float*)d_in[1];
  const float* b1h = (const float*)d_in[2];
  const float* W2h = (const float*)d_in[3];
  const float* b2h = (const float*)d_in[4];
  const float* W1y = (const float*)d_in[5];
  const float* b1y = (const float*)d_in[6];
  const float* W2y = (const float*)d_in[7];
  const float* b2y = (const float*)d_in[8];

  unsigned short* ws  = (unsigned short*)d_ws;
  unsigned short* w1h = ws;                               // 320*512 = 163840
  unsigned short* w2h = ws + 163840;                      // 512*256 = 131072
  unsigned short* w1y = ws + 163840 + 131072;             // 256*256 =  65536
  unsigned short* w2y = ws + 163840 + 131072 + 65536;     // 256*64  =  16384

  swizzle_w<<<640, 256, 0, stream>>>(W1h, w1h, 320, 512);
  swizzle_w<<<512, 256, 0, stream>>>(W2h, w2h, 512, 256);
  swizzle_w<<<256, 256, 0, stream>>>(W1y, w1y, 256, 256);
  swizzle_w<<<64,  256, 0, stream>>>(W2y, w2y, 256, 64);

  float* outs = (float*)d_out;
  float* hid  = outs + (size_t)B_SZ * S_SZ * DOUT;

  // 8 workgroups (32 batch rows each), 8 waves, 53248 B dynamic LDS.
  rnn_persistent<<<8, 256, 53248, stream>>>(x, w1h, w2h, w1y, w2y,
                                            b1h, b2h, b1y, b2y, outs, hid);
}
